// MANN_Attn_78572131713092
// MI455X (gfx1250) — matrix operand order repaired and checked by reading, not compiled, not run on hardware
//
#include <hip/hip_runtime.h>

typedef __attribute__((ext_vector_type(16))) _Float16 v16h;
typedef __attribute__((ext_vector_type(8)))  _Float16 v8h;
typedef __attribute__((ext_vector_type(8)))  float    v8f;

// ---------------- problem constants ----------------
constexpr int BQ   = 64;           // batch
constexpr int NC   = 16;           // NCELLS
constexpr int TT   = 16, LLn = 32, LDm = 24;
constexpr int TLn  = TT * LLn;     // 512 encoder steps
constexpr int SDm  = TT * (LDm + 1); // 400 decoder steps
constexpr int OUTD = 153;
constexpr int NTHR = 256;
constexpr int NWAVE = 8;

// ---------------- output layout (flat concat of the 4-tuple) ----------------
constexpr size_t OO_PRE = 0;                       // (B,SD,153)
constexpr size_t OO_LAB = 3916800;                 // (B,SD)
constexpr size_t OO_PT  = 3942400;                 // (B,25,153)
constexpr size_t OO_LT  = 4187200;                 // (B,25)

// ---------------- workspace layout ----------------
constexpr size_t ALGN(size_t x){ return (x + 255) & ~size_t(255); }
constexpr size_t OFF_EWX0 = 0;
constexpr size_t OFF_EWX1 = OFF_EWX0 + ALGN(128*192*2);
constexpr size_t OFF_EWH0 = OFF_EWX1 + ALGN(128*192*2);
constexpr size_t OFF_EWH1 = OFF_EWH0 + ALGN(64*192*2);
constexpr size_t OFF_EWI0 = OFF_EWH1 + ALGN(64*192*2);
constexpr size_t OFF_EWI1 = OFF_EWI0 + ALGN(64*272*2);
constexpr size_t OFF_EBI0 = OFF_EWI1 + ALGN(64*272*2);
constexpr size_t OFF_EBI1 = OFF_EBI0 + ALGN(272*4);
constexpr size_t OFF_DWX  = OFF_EBI1 + ALGN(272*4);
constexpr size_t OFF_DWH  = OFF_DWX + ALGN(256*384*2);
constexpr size_t OFF_DWI  = OFF_DWH + ALGN(128*384*2);
constexpr size_t OFF_DBI  = OFF_DWI + ALGN(128*528*2);
constexpr size_t OFF_CW   = OFF_DBI + ALGN(528*4);
constexpr size_t OFF_OW   = OFF_CW  + ALGN(192*128*2);
constexpr size_t OFF_OB   = OFF_OW  + ALGN(256*160*2);
constexpr size_t OFF_MENC0= OFF_OB  + ALGN(160*4);
constexpr size_t OFF_MENC1= OFF_MENC0 + ALGN((size_t)BQ*NC*64*4);
constexpr size_t OFF_MDEC = OFF_MENC1 + ALGN((size_t)BQ*NC*64*4);
constexpr size_t OFF_HS0  = OFF_MDEC + ALGN((size_t)BQ*NC*128*4);
constexpr size_t OFF_HS1  = OFF_HS0 + ALGN((size_t)TLn*BQ*64*4);

// ---------------- LDS layouts (dynamic shared) ----------------
constexpr int ESM_H=0, ESM_AH=16384, ESM_A=24576, ESM_SCR=40960,
              ESM_SC=110592, ESM_WW=114688, ESM_ST=118784, ESM_TOT=119808;
constexpr int DSM_H=0, DSM_AH=32768, DSM_AG=49152, DSM_AC=81920,
              DSM_C1=114688, DSM_C2=122880, DSM_SCR=131072,
              DSM_SC=266240, DSM_WW=270336, DSM_ST=274432,
              DSM_AS=275456, DSM_HD=283648, DSM_TOT=283904;

__device__ __forceinline__ float sigm(float x){ return 1.f / (1.f + __expf(-x)); }
__device__ __forceinline__ float softp(float x){ return (x > 20.f) ? x : log1pf(__expf(x)); }

// ---------------- WMMA fragment loaders ----------------
// A (LDS, row-major f16): ISA 16-bit A 16x32 layout -> two aligned 16B chunks per lane.
__device__ __forceinline__ v16h ldA(const _Float16* A, int lda, int m0, int k0, int lane){
  int row = m0 + (lane & 15);
  int ko  = (lane >= 16) ? 8 : 0;
  const v8h* p = (const v8h*)(A + row * lda + k0 + ko);   // 16B aligned: lda%8==0, k0%32==0
  v8h lo = p[0];     // K = k0+ko + 0..7
  v8h hi = p[2];     // K = k0+ko + 16..23
  return __builtin_shufflevector(lo, hi, 0,1,2,3,4,5,6,7,8,9,10,11,12,13,14,15);
}

// C[BQ x N] (+)= A[BQ x K] (LDS f16) @ W (global f16, pre-swizzled fragments) + bias
// W fragment storage: frag[( (ntOff+nt)*KT + kt )*32 + lane] = v16h (32B contiguous per lane)
__device__ __forceinline__ void mm(const _Float16* A, int lda,
                                   const _Float16* W, int ntOff,
                                   float* C, int ldc,
                                   const float* bias, bool accum,
                                   int M, int N, int K, int tid){
  const v16h* WF = (const v16h*)W;
  int wave = tid >> 5, lane = tid & 31;
  int mt = M >> 4, nt = N >> 4, KT = K >> 5;
  for (int t = wave; t < mt * nt; t += NWAVE){
    int m0  = (t / nt) << 4;
    int n0t = (t % nt);
    int col = (n0t << 4) + (lane & 15);
    int rb  = m0 + ((lane >= 16) ? 8 : 0);
    float bv = bias ? bias[col] : 0.f;
    v8f c;
#pragma unroll
    for (int j = 0; j < 8; ++j) c[j] = accum ? (C[(rb + j) * ldc + col] + bv) : bv;
    const v16h* wp = WF + ((size_t)(ntOff + n0t) * KT) * 32 + lane;
    for (int kt = 0; kt < KT; ++kt){
      v16h a = ldA(A, lda, m0, kt << 5, lane);
      v16h b = wp[(size_t)kt * 32];
      c = __builtin_amdgcn_wmma_f32_16x16x32_f16(false, a, false, b, (short)0, c, false, false);
    }
#pragma unroll
    for (int j = 0; j < 8; ++j) C[(rb + j) * ldc + col] = c[j];
  }
}

// ---------------- DNC memory (cosine addressing + write + read) ----------------
template<int W>
__device__ void dnc_mem(const float* __restrict__ ifc, int ldi,
                        float* __restrict__ M,
                        _Float16* __restrict__ rdst, int ldr,
                        float* sc, float* ww, float* st, int tid){
  // --- write head: k_w = cols [W,2W), beta_w = 4W+1, g_w = 4W+2 ---
  if (tid < BQ){
    int b = tid; float s = 0.f;
    for (int w = 0; w < W; ++w){ float v = ifc[b*ldi + W + w]; s += v*v; }
    st[b]      = sqrtf(s);
    st[64 + b] = softp(ifc[b*ldi + 4*W + 1]) + 1.f;
    st[128+ b] = sigm (ifc[b*ldi + 4*W + 2]);
  }
  __syncthreads();
  for (int i = tid; i < BQ*NC; i += NTHR){
    int b = i / NC, n = i % NC;
    const float* m = M + ((size_t)b*NC + n) * W;
    float num = 0.f, nn = 0.f;
    for (int w = 0; w < W; ++w){ float mv = m[w]; num += mv * ifc[b*ldi + W + w]; nn += mv*mv; }
    sc[i] = st[64 + b] * num / (sqrtf(nn) * st[b] + 1e-6f);
  }
  __syncthreads();
  if (tid < BQ){
    int b = tid; float mx = -1e30f;
    for (int n = 0; n < NC; ++n) mx = fmaxf(mx, sc[b*NC + n]);
    float sm = 0.f;
    for (int n = 0; n < NC; ++n){ float e = __expf(sc[b*NC + n] - mx); sc[b*NC + n] = e; sm += e; }
    float g = st[128 + b] / sm;
    for (int n = 0; n < NC; ++n) ww[b*NC + n] = g * sc[b*NC + n];
  }
  __syncthreads();
  // --- memory update: erase cols [2W,3W), write vec cols [3W,4W) ---
  for (int i = tid; i < BQ*NC*W; i += NTHR){
    int b = i / (NC*W), rem = i % (NC*W), n = rem / W, w = rem % W;
    float wwv = ww[b*NC + n];
    float er  = sigm(ifc[b*ldi + 2*W + w]);
    float wv  =      ifc[b*ldi + 3*W + w];
    M[i] = M[i] * (1.f - wwv*er) + wwv*wv;
  }
  __threadfence();
  __syncthreads();
  // --- read head: k_r = cols [0,W), beta_r = 4W ---
  if (tid < BQ){
    int b = tid; float s = 0.f;
    for (int w = 0; w < W; ++w){ float v = ifc[b*ldi + w]; s += v*v; }
    st[b]      = sqrtf(s);
    st[64 + b] = softp(ifc[b*ldi + 4*W]) + 1.f;
  }
  __syncthreads();
  for (int i = tid; i < BQ*NC; i += NTHR){
    int b = i / NC, n = i % NC;
    const float* m = M + ((size_t)b*NC + n) * W;
    float num = 0.f, nn = 0.f;
    for (int w = 0; w < W; ++w){ float mv = m[w]; num += mv * ifc[b*ldi + w]; nn += mv*mv; }
    sc[i] = st[64 + b] * num / (sqrtf(nn) * st[b] + 1e-6f);
  }
  __syncthreads();
  if (tid < BQ){
    int b = tid; float mx = -1e30f;
    for (int n = 0; n < NC; ++n) mx = fmaxf(mx, sc[b*NC + n]);
    float sm = 0.f;
    for (int n = 0; n < NC; ++n){ float e = __expf(sc[b*NC + n] - mx); sc[b*NC + n] = e; sm += e; }
    float inv = 1.f / sm;
    for (int n = 0; n < NC; ++n) ww[b*NC + n] = sc[b*NC + n] * inv;
  }
  __syncthreads();
  for (int i = tid; i < BQ*W; i += NTHR){
    int b = i / W, w = i % W; float acc = 0.f;
#pragma unroll
    for (int n = 0; n < NC; ++n) acc += ww[b*NC + n] * M[((size_t)b*NC + n) * W + w];
    rdst[b*ldr + w] = (_Float16)acc;
  }
  __syncthreads();
}

// ---------------- weight convert: f32 row-major -> f16 WMMA-B fragment order ----------------
// dst index decomposition: i = ((nt*KT + kt)*32 + lane)*16 + elem
// lane 0..15  hold K = kt*32 + 0..7, 16..23
// lane 16..31 hold K = kt*32 + 8..15, 24..31
__global__ void cvt_swz(const float* __restrict__ src, _Float16* __restrict__ dst,
                        int K, int Ns, int Nd){
  int i = blockIdx.x * blockDim.x + threadIdx.x;
  if (i >= K * Nd) return;
  int elem = i & 15;
  int lane = (i >> 4) & 31;
  int fi   = i >> 9;
  int KT   = K >> 5;
  int kt = fi % KT, nt = fi / KT;
  int row = (kt << 5) + ((lane >= 16) ? 8 : 0) + (elem & 7) + ((elem & 8) ? 16 : 0);
  int col = (nt << 4) + (lane & 15);
  dst[i] = (col < Ns) ? (_Float16)src[row * Ns + col] : (_Float16)0.f;
}
__global__ void pad_bias(const float* __restrict__ src, float* __restrict__ dst, int Ns, int Nd){
  int i = blockIdx.x * blockDim.x + threadIdx.x;
  if (i < Nd) dst[i] = (i < Ns) ? src[i] : 0.f;
}

// ---------------- labels ----------------
__global__ void labels_kernel(const int* __restrict__ targets, float* __restrict__ out){
  int i = blockIdx.x * blockDim.x + threadIdx.x;
  if (i >= BQ * SDm) return;
  int b = i / SDm, s = i % SDm;
  int tt = s / (LDm + 1), j = s % (LDm + 1);
  float v = (j < LDm) ? (float)targets[(b*TT + tt)*LDm + j] : ((tt == TT-1) ? 152.f : 151.f);
  out[OO_LAB + i] = v;
  if (s >= SDm - (LDm + 1)) out[OO_LT + (size_t)b*(LDm+1) + (s - (SDm - (LDm+1)))] = v;
}

// ---------------- persistent encoder: one workgroup per encoder ----------------
__global__ __launch_bounds__(NTHR) void enc_kernel(
    const int* __restrict__ codes1, const int* __restrict__ codes2,
    const float* __restrict__ emb0, const float* __restrict__ emb1,
    const float* __restrict__ bx0, const float* __restrict__ bh0,
    const float* __restrict__ bx1, const float* __restrict__ bh1,
    char* __restrict__ ws)
{
  extern __shared__ char smem[];
  const int tid = threadIdx.x;
  const int enc = blockIdx.x;
  const int*   codes = enc ? codes2 : codes1;
  const float* emb   = enc ? emb1   : emb0;
  const float* bx    = enc ? bx1    : bx0;
  const float* bh    = enc ? bh1    : bh0;
  const _Float16* wx = (const _Float16*)(ws + (enc ? OFF_EWX1 : OFF_EWX0));
  const _Float16* wh = (const _Float16*)(ws + (enc ? OFF_EWH1 : OFF_EWH0));
  const _Float16* wi = (const _Float16*)(ws + (enc ? OFF_EWI1 : OFF_EWI0));
  const float* bip   = (const float*)(ws + (enc ? OFF_EBI1 : OFF_EBI0));
  float* M  = (float*)(ws + (enc ? OFF_MENC1 : OFF_MENC0));
  float* hs = (float*)(ws + (enc ? OFF_HS1   : OFF_HS0));

  float*    h   = (float*)(smem + ESM_H);      // B x 64 f32
  _Float16* Ah  = (_Float16*)(smem + ESM_AH);  // B x 64 f16 (h)
  _Float16* A   = (_Float16*)(smem + ESM_A);   // B x 128 f16 ([x | r])
  float*    scr = (float*)(smem + ESM_SCR);    // G(B x192) + HN(B x64) | iface(B x272)
  float*    G   = scr;
  float*    HN  = scr + BQ * 192;
  float*    sc  = (float*)(smem + ESM_SC);
  float*    ww  = (float*)(smem + ESM_WW);
  float*    st  = (float*)(smem + ESM_ST);

  for (int i = tid; i < BQ*64;  i += NTHR){ h[i] = 0.f; Ah[i] = (_Float16)0.f; }
  for (int i = tid; i < BQ*128; i += NTHR) A[i] = (_Float16)0.f;
  for (int i = tid; i < BQ*NC*64; i += NTHR) M[i] = 0.f;
  __threadfence();
  __syncthreads();

  for (int t = 0; t < TLn; ++t){
    // gather x_t into A[:,0:64]
    for (int i = tid; i < BQ*64; i += NTHR){
      int b = i >> 6, e = i & 63;
      int code = codes[b*TLn + t];
      A[b*128 + e] = (_Float16)emb[code*64 + e];
    }
    __syncthreads();
    mm(A, 128, wx, 0, G, 192, bx, false, BQ, 192, 128, tid);   // gx = [x|r]@wx + bx
    __syncthreads();
    mm(Ah, 64, wh, 0, G,  192, bh,       true,  BQ, 128, 64, tid); // + h@wh[:,:128]+bh
    mm(Ah, 64, wh, 8, HN, 64,  bh + 128, false, BQ, 64,  64, tid); // hn = h@wh[:,128:]+bh
    __syncthreads();
    for (int i = tid; i < BQ*64; i += NTHR){                     // GRU gates
      int b = i >> 6, e = i & 63;
      float rg = sigm(G[b*192 + e]);
      float z  = sigm(G[b*192 + 64 + e]);
      float n  = tanhf(G[b*192 + 128 + e] + rg * HN[i]);
      float hn = (1.f - z) * n + z * h[i];
      h[i] = hn; Ah[i] = (_Float16)hn;
      hs[(size_t)t*BQ*64 + i] = hn;
    }
    __syncthreads();
    mm(Ah, 64, wi, 0, scr, 272, bip, false, BQ, 272, 64, tid); // iface (padded 259->272)
    __syncthreads();
    dnc_mem<64>(scr, 272, M, A + 64, 128, sc, ww, st, tid);    // writes r -> A[:,64:128]
  }
}

// ---------------- persistent decoder ----------------
__global__ __launch_bounds__(NTHR) void dec_kernel(
    const int* __restrict__ targets, const float* __restrict__ emb2,
    const float* __restrict__ bx, const float* __restrict__ bh,
    const float* __restrict__ attn0, const float* __restrict__ attn1,
    const float* __restrict__ comb_b,
    char* __restrict__ ws, float* __restrict__ out)
{
  extern __shared__ char smem[];
  const int tid = threadIdx.x;
  const _Float16* wx  = (const _Float16*)(ws + OFF_DWX);
  const _Float16* wh  = (const _Float16*)(ws + OFF_DWH);
  const _Float16* wi  = (const _Float16*)(ws + OFF_DWI);
  const float*    bip = (const float*)(ws + OFF_DBI);
  const _Float16* cw  = (const _Float16*)(ws + OFF_CW);
  const _Float16* ow  = (const _Float16*)(ws + OFF_OW);
  const float*    obp = (const float*)(ws + OFF_OB);
  float* M = (float*)(ws + OFF_MDEC);
  const float* hs1 = (const float*)(ws + OFF_HS0);
  const float* hs2 = (const float*)(ws + OFF_HS1);

  float*    h   = (float*)(smem + DSM_H);     // B x 128
  _Float16* Ah  = (_Float16*)(smem + DSM_AH); // B x 128 f16
  _Float16* AG  = (_Float16*)(smem + DSM_AG); // B x 256 f16 ([comb | r])
  _Float16* AC  = (_Float16*)(smem + DSM_AC); // B x 192/256 f16 ([tok|c1|c2] then [h2|c1|c2])
  _Float16* c1h = (_Float16*)(smem + DSM_C1);
  _Float16* c2h = (_Float16*)(smem + DSM_C2);
  float*    scr = (float*)(smem + DSM_SCR);   // G(Bx384)+HN(Bx128) | iface(Bx528) | C bufs
  float*    G   = scr;
  float*    HN  = scr + BQ * 384;
  float*    sc  = (float*)(smem + DSM_SC);
  float*    ww  = (float*)(smem + DSM_WW);
  float*    st  = (float*)(smem + DSM_ST);
  float*    as  = (float*)(smem + DSM_AS);
  float*    hd  = (float*)(smem + DSM_HD);

  // init: h = concat(hs1[L-1], hs2[L-1]); r = 0; M = 0
  for (int i = tid; i < BQ*128; i += NTHR){
    int b = i >> 7, e = i & 127;
    float v = (e < 64) ? hs1[((size_t)(LLn-1)*BQ + b)*64 + e]
                       : hs2[((size_t)(LLn-1)*BQ + b)*64 + (e - 64)];
    h[i] = v; Ah[i] = (_Float16)v;
  }
  for (int i = tid; i < BQ*256; i += NTHR) AG[i] = (_Float16)0.f;
  for (int i = tid; i < BQ*NC*128; i += NTHR) M[i] = 0.f;
  __threadfence();
  __syncthreads();

  auto attend = [&](const float* hsrc, const float* aw, _Float16* cdst, int t){
    for (int b = tid; b < BQ; b += NTHR){
      float s = 0.f;
      for (int e = 0; e < 128; ++e) s += h[b*128 + e] * aw[e];
      hd[b] = s;
    }
    __syncthreads();
    for (int i = tid; i < BQ*LLn; i += NTHR){
      int b = i / LLn, l = i % LLn;
      const float* o = hsrc + ((size_t)(t*LLn + l)*BQ + b) * 64;
      float s = hd[b];
      for (int e = 0; e < 64; ++e) s += o[e] * aw[128 + e];
      as[i] = s;
    }
    __syncthreads();
    for (int b = tid; b < BQ; b += NTHR){
      float mx = -1e30f;
      for (int l = 0; l < LLn; ++l) mx = fmaxf(mx, as[b*LLn + l]);
      float sm = 0.f;
      for (int l = 0; l < LLn; ++l){ float e = __expf(as[b*LLn + l] - mx); as[b*LLn + l] = e; sm += e; }
      float inv = 1.f / sm;
      for (int l = 0; l < LLn; ++l) as[b*LLn + l] *= inv;
    }
    __syncthreads();
    for (int i = tid; i < BQ*64; i += NTHR){
      int b = i >> 6, e = i & 63; float acc = 0.f;
      for (int l = 0; l < LLn; ++l) acc += as[b*LLn + l] * hsrc[((size_t)(t*LLn + l)*BQ + b)*64 + e];
      cdst[i] = (_Float16)acc;
    }
    __syncthreads();
  };

  for (int s = 0; s < SDm; ++s){
    int tt = s / (LDm + 1), j = s % (LDm + 1);
    attend(hs1, attn0, c1h, tt);
    attend(hs2, attn1, c2h, tt);
    // build Acomb = [tok_emb | c1 | c2]
    for (int i = tid; i < BQ*64; i += NTHR){
      int b = i >> 6, e = i & 63;
      int id = (j == 0) ? ((tt == 0) ? 150 : 151) : targets[(b*TT + tt)*LDm + (j - 1)];
      AC[b*192 + e]        = (_Float16)emb2[id*64 + e];
      AC[b*192 + 64 + e]   = c1h[i];
      AC[b*192 + 128 + e]  = c2h[i];
    }
    __syncthreads();
    mm(AC, 192, cw, 0, scr, 128, comb_b, false, BQ, 128, 192, tid);  // comb pre-act
    __syncthreads();
    for (int i = tid; i < BQ*128; i += NTHR){                        // relu -> AG[:,0:128]
      int b = i >> 7, e = i & 127;
      AG[b*256 + e] = (_Float16)fmaxf(scr[i], 0.f);
    }
    __syncthreads();
    mm(AG, 256, wx, 0, G, 384, bx, false, BQ, 384, 256, tid);        // gx
    __syncthreads();
    mm(Ah, 128, wh, 0,  G,  384, bh,       true,  BQ, 256, 128, tid);
    mm(Ah, 128, wh, 16, HN, 128, bh + 256, false, BQ, 128, 128, tid);
    __syncthreads();
    for (int i = tid; i < BQ*128; i += NTHR){                        // GRU
      int b = i >> 7, e = i & 127;
      float rg = sigm(G[b*384 + e]);
      float z  = sigm(G[b*384 + 128 + e]);
      float n  = tanhf(G[b*384 + 256 + e] + rg * HN[i]);
      float hn = (1.f - z) * n + z * h[i];
      h[i] = hn; Ah[i] = (_Float16)hn;
    }
    __syncthreads();
    mm(Ah, 128, wi, 0, scr, 528, bip, false, BQ, 528, 128, tid);     // iface (515->528)
    __syncthreads();
    dnc_mem<128>(scr, 528, M, AG + 128, 256, sc, ww, st, tid);       // r -> AG[:,128:256]
    // build Aout = [h2 | c1 | c2] (aliases AC)
    for (int i = tid; i < BQ*128; i += NTHR){ int b = i >> 7, e = i & 127; AC[b*256 + e] = Ah[i]; }
    for (int i = tid; i < BQ*64; i += NTHR){
      int b = i >> 6, e = i & 63;
      AC[b*256 + 128 + e] = c1h[i];
      AC[b*256 + 192 + e] = c2h[i];
    }
    __syncthreads();
    mm(AC, 256, ow, 0, scr, 160, obp, false, BQ, 160, 256, tid);     // logits (153->160)
    __syncthreads();
    for (int i = tid; i < BQ*OUTD; i += NTHR){
      int b = i / OUTD, k = i % OUTD;
      float v = scr[b*160 + k];
      out[OO_PRE + ((size_t)b*SDm + s)*OUTD + k] = v;
      if (s >= SDm - (LDm + 1))
        out[OO_PT + ((size_t)b*(LDm+1) + (s - (SDm - (LDm+1))))*OUTD + k] = v;
    }
    __syncthreads();
  }
}

// ---------------- host launch ----------------
extern "C" void kernel_launch(void* const* d_in, const int* in_sizes, int n_in,
                              void* d_out, int out_size, void* d_ws, size_t ws_size,
                              hipStream_t stream){
  (void)in_sizes; (void)n_in; (void)out_size; (void)ws_size;
  char*  ws  = (char*)d_ws;
  float* out = (float*)d_out;

  (void)hipFuncSetAttribute((const void*)enc_kernel,
                            hipFuncAttributeMaxDynamicSharedMemorySize, ESM_TOT);
  (void)hipFuncSetAttribute((const void*)dec_kernel,
                            hipFuncAttributeMaxDynamicSharedMemorySize, DSM_TOT);

  auto cvt = [&](const void* src, size_t off, int K, int Ns, int Nd){
    int n = K * Nd;
    cvt_swz<<<(n + 255) / 256, 256, 0, stream>>>((const float*)src, (_Float16*)(ws + off), K, Ns, Nd);
  };
  auto pbias = [&](const void* src, size_t off, int Ns, int Nd){
    pad_bias<<<(Nd + 255) / 256, 256, 0, stream>>>((const float*)src, (float*)(ws + off), Ns, Nd);
  };

  cvt(d_in[6],  OFF_EWX0, 128, 192, 192);
  cvt(d_in[7],  OFF_EWH0,  64, 192, 192);
  cvt(d_in[10], OFF_EWI0,  64, 259, 272);
  cvt(d_in[12], OFF_EWX1, 128, 192, 192);
  cvt(d_in[13], OFF_EWH1,  64, 192, 192);
  cvt(d_in[16], OFF_EWI1,  64, 259, 272);
  cvt(d_in[18], OFF_DWX,  256, 384, 384);
  cvt(d_in[19], OFF_DWH,  128, 384, 384);
  cvt(d_in[22], OFF_DWI,  128, 515, 528);
  cvt(d_in[26], OFF_CW,   192, 128, 128);
  cvt(d_in[28], OFF_OW,   256, 153, 160);
  pbias(d_in[11], OFF_EBI0, 259, 272);
  pbias(d_in[17], OFF_EBI1, 259, 272);
  pbias(d_in[23], OFF_DBI,  515, 528);
  pbias(d_in[29], OFF_OB,   153, 160);

  labels_kernel<<<(BQ*SDm + 255)/256, 256, 0, stream>>>((const int*)d_in[2], out);

  enc_kernel<<<2, NTHR, ESM_TOT, stream>>>(
      (const int*)d_in[0], (const int*)d_in[1],
      (const float*)d_in[3], (const float*)d_in[4],
      (const float*)d_in[8],  (const float*)d_in[9],
      (const float*)d_in[14], (const float*)d_in[15],
      ws);

  dec_kernel<<<1, NTHR, DSM_TOT, stream>>>(
      (const int*)d_in[2], (const float*)d_in[5],
      (const float*)d_in[20], (const float*)d_in[21],
      (const float*)d_in[24], (const float*)d_in[25],
      (const float*)d_in[27],
      ws, out);
}
